// Decoder_3075196583967
// MI455X (gfx1250) — matrix operand order repaired and checked by reading, not compiled, not run on hardware
//
#include <hip/hip_runtime.h>
#include <math.h>

// Problem dimensions (match reference)
#define DD   512            // model size
#define VV   32000          // vocab
#define TQn  128            // decoder steps
#define TKn  128            // encoder length
#define BBn  16             // batch
#define G4   (4*DD)         // 2048 gate width
#define MR   (TQn*BBn)      // 2048 flattened (t,b) rows

typedef __attribute__((ext_vector_type(16))) __bf16 v16bf;
typedef __attribute__((ext_vector_type(8)))  float  v8f;

union FragU { unsigned int u[8]; v16bf v; };

__device__ __forceinline__ unsigned short f2bf(float f) {
  unsigned int u = __builtin_bit_cast(unsigned int, f);
  u += 0x7FFFu + ((u >> 16) & 1u);          // round-to-nearest-even
  return (unsigned short)(u >> 16);
}

// A fragment: 16x32 bf16, row m = lane&15, half h = lane>>4.
// VGPR v holds K = 8h + 2v (v<4) or 8h + 16 + 2(v-4) (v>=4).
// Lowered by the compiler to two global_load_b128 per fragment.
__device__ __forceinline__ v16bf load_a_frag(const unsigned short* A, int K,
                                             int k0, int lane) {
  int m = lane & 15, h = lane >> 4;
  const unsigned short* p = A + (size_t)m * K + k0 + 8 * h;
  FragU f;
#pragma unroll
  for (int v = 0; v < 8; ++v) {
    int off = (v < 4) ? (2 * v) : (2 * v + 8);
    f.u[v] = *(const unsigned int*)(p + off);
  }
  return f.v;
}

// B fragment (K x N = 32x16) built from row-major W[N,K]: B[k][n] = W[n][k].
// col n = lane&15, half h = lane>>4; VGPR v holds K = 8h + 2v (v<4) or 8h + 16 + 2(v-4) (v>=4).
__device__ __forceinline__ v16bf load_b_frag(const unsigned short* W, int K,
                                             int k0, int lane) {
  int n = lane & 15, h = lane >> 4;
  const unsigned short* p = W + (size_t)n * K + k0 + 8 * h;
  FragU f;
#pragma unroll
  for (int v = 0; v < 8; ++v) f.u[v] = *(const unsigned int*)(p + ((v < 4) ? (2 * v) : (2 * v + 8)));
  return f.v;
}

// ---------------------------------------------------------------------------
// Simple GEMM: one 16x16 tile per wave, supports fused Cin add (used for the
// recurrent gate GEMM where M = B = 16 and Cin = xg[t]).
// C[M,N] = A[M,K](bf16) @ W[N,K]^T(bf16) + Cin(opt,[M,N]) + bias(opt,[N])
// ---------------------------------------------------------------------------
__global__ void wmma_gemm(const unsigned short* __restrict__ A,
                          const unsigned short* __restrict__ W,
                          const float* __restrict__ Cin,
                          const float* __restrict__ bias,
                          float* __restrict__ C,
                          int M, int N, int K, int total_tiles) {
  int wave = threadIdx.x >> 5;
  int lane = threadIdx.x & 31;
  int tile = blockIdx.x * 4 + wave;
  if (tile >= total_tiles) return;
  int ntn = N >> 4;
  int tm = tile / ntn, tn = tile - tm * ntn;
  const unsigned short* Ab = A + (size_t)(tm * 16) * K;
  const unsigned short* Wt = W + (size_t)(tn * 16) * K;
  v8f acc = {};
  for (int k0 = 0; k0 < K; k0 += 32) {
    v16bf a = load_a_frag(Ab, K, k0, lane);
    v16bf b = load_b_frag(Wt, K, k0, lane);
    acc = __builtin_amdgcn_wmma_f32_16x16x32_bf16(false, a, false, b,
                                                  (short)0, acc, false, false);
  }
  int n = lane & 15, h = lane >> 4;
  int col = tn * 16 + n;
  float bv = bias ? bias[col] : 0.0f;
#pragma unroll
  for (int r = 0; r < 8; ++r) {
    int row = tm * 16 + r + 8 * h;
    size_t idx = (size_t)row * N + col;
    float v = acc[r] + bv;
    if (Cin) v += Cin[idx];
    C[idx] = v;
  }
}

// ---------------------------------------------------------------------------
// Register-blocked GEMM: each wave computes a 32x64 region (2 M-tiles x 4
// N-tiles). Per k-step: 2 A-frags + 4 B-frags feed 8 WMMAs -> 1.5 b128 loads
// per WMMA (vs 4 for the naive kernel), cutting L2 fragment traffic ~2.7x on
// the 2048x32000x512 logits GEMM whose 32MB bf16 weight lives in the 192MB L2.
// Requires M%32==0, N%64==0, K%32==0 (true for all large GEMMs here).
// ---------------------------------------------------------------------------
__global__ void wmma_gemm_blk(const unsigned short* __restrict__ A,
                              const unsigned short* __restrict__ W,
                              const float* __restrict__ bias,
                              float* __restrict__ C,
                              int M, int N, int K, int total_blks) {
  int wave = threadIdx.x >> 5;
  int lane = threadIdx.x & 31;
  int blk = blockIdx.x * 4 + wave;
  if (blk >= total_blks) return;
  int nbn = N >> 6;                       // 64-wide column blocks
  int bm = blk / nbn, bn = blk - bm * nbn;
  const unsigned short* A0 = A + (size_t)(bm * 32) * K;
  const unsigned short* A1 = A0 + (size_t)16 * K;
  const unsigned short* W0 = W + (size_t)(bn * 64) * K;
  v8f acc[2][4];
#pragma unroll
  for (int i = 0; i < 2; ++i)
#pragma unroll
    for (int j = 0; j < 4; ++j) acc[i][j] = (v8f){};
  for (int k0 = 0; k0 < K; k0 += 32) {
    v16bf a0 = load_a_frag(A0, K, k0, lane);
    v16bf a1 = load_a_frag(A1, K, k0, lane);
    v16bf b[4];
#pragma unroll
    for (int j = 0; j < 4; ++j)
      b[j] = load_b_frag(W0 + (size_t)(16 * j) * K, K, k0, lane);
#pragma unroll
    for (int j = 0; j < 4; ++j) {
      acc[0][j] = __builtin_amdgcn_wmma_f32_16x16x32_bf16(
          false, a0, false, b[j], (short)0, acc[0][j], false, false);
      acc[1][j] = __builtin_amdgcn_wmma_f32_16x16x32_bf16(
          false, a1, false, b[j], (short)0, acc[1][j], false, false);
    }
  }
  int n = lane & 15, h = lane >> 4;
#pragma unroll
  for (int i = 0; i < 2; ++i) {
    int row_base = bm * 32 + i * 16 + 8 * h;
#pragma unroll
    for (int j = 0; j < 4; ++j) {
      int col = bn * 64 + 16 * j + n;
      float bv = bias ? bias[col] : 0.0f;
#pragma unroll
      for (int r = 0; r < 8; ++r)
        C[(size_t)(row_base + r) * N + col] = acc[i][j][r] + bv;
    }
  }
}

__global__ void cvt_bf16_kernel(const float* __restrict__ s,
                                unsigned short* __restrict__ d, int n) {
  int i = blockIdx.x * blockDim.x + threadIdx.x;
  if (i < n) d[i] = f2bf(s[i]);
}

__global__ void add2_kernel(const float* __restrict__ a,
                            const float* __restrict__ b,
                            float* __restrict__ o, int n) {
  int i = blockIdx.x * blockDim.x + threadIdx.x;
  if (i < n) o[i] = a[i] + b[i];
}

__global__ void embed_kernel(const int* __restrict__ inputs,
                             const float* __restrict__ emb,
                             float* __restrict__ x, int total) {
  int i = blockIdx.x * blockDim.x + threadIdx.x;
  if (i >= total) return;
  int tb = i / DD, d = i - tb * DD;
  x[i] = emb[(size_t)inputs[tb] * DD + d];
}

__global__ void init_hc_kernel(const float* __restrict__ h0,
                               const float* __restrict__ c0, int layer,
                               float* __restrict__ h, float* __restrict__ c,
                               unsigned short* __restrict__ hbf) {
  int i = blockIdx.x * blockDim.x + threadIdx.x;
  if (i >= BBn * DD) return;
  float hv = h0[(size_t)layer * BBn * DD + i];
  h[i] = hv;
  hbf[i] = f2bf(hv);
  c[i] = c0[(size_t)layer * BBn * DD + i];
}

// Gate nonlinearity + state update. G is [B, 4D] with PyTorch order i,f,g,o.
// yout[t,b,d] = h_new (+ skip[t,b,d] if skip != nullptr; in place = residual).
__global__ void lstm_point_kernel(const float* __restrict__ G,
                                  float* __restrict__ h, float* __restrict__ c,
                                  unsigned short* __restrict__ hbf,
                                  float* __restrict__ yout,
                                  const float* __restrict__ skip, int t) {
  int i = blockIdx.x * blockDim.x + threadIdx.x;
  if (i >= BBn * DD) return;
  int b = i / DD, d = i - b * DD;
  const float* g = G + (size_t)b * G4;
  float gi = g[d], gf = g[DD + d], gg = g[2 * DD + d], go = g[3 * DD + d];
  float si = 1.0f / (1.0f + expf(-gi));
  float sf = 1.0f / (1.0f + expf(-gf));
  float so = 1.0f / (1.0f + expf(-go));
  float cn = sf * c[i] + si * tanhf(gg);
  float hn = so * tanhf(cn);
  c[i] = cn;
  h[i] = hn;
  hbf[i] = f2bf(hn);
  size_t yi = ((size_t)t * BBn + b) * DD + d;
  yout[yi] = skip ? (hn + skip[yi]) : hn;
}

__global__ void copy_hc_kernel(const float* __restrict__ h,
                               const float* __restrict__ c,
                               float* __restrict__ oh, float* __restrict__ oc) {
  int i = blockIdx.x * blockDim.x + threadIdx.x;
  if (i >= BBn * DD) return;
  oh[i] = h[i];
  oc[i] = c[i];
}

// Bahdanau attention fused: scores (tanh inside -> not a GEMM), masked
// softmax, context. One block per (q,b), blockDim = TK (128, 4 waves).
__global__ void attn_kernel(const float* __restrict__ qp,
                            const float* __restrict__ kp,
                            const float* __restrict__ vp,
                            const float* __restrict__ mask,
                            const float* __restrict__ Ww,
                            const float* __restrict__ Wb,
                            float* __restrict__ ctx) {
  __shared__ float qs[DD];
  __shared__ float e[TKn];
  __shared__ float red[TKn];
  int qb = blockIdx.x;
  int q = qb / BBn, b = qb - q * BBn;
  int tid = threadIdx.x;
  for (int d = tid; d < DD; d += TKn)
    qs[d] = qp[((size_t)q * BBn + b) * DD + d];
  __syncthreads();
  int k = tid;  // one thread per encoder position
  const float* krow = kp + ((size_t)k * BBn + b) * DD;
  float s = 0.0f;
  for (int d = 0; d < DD; ++d) s += tanhf(qs[d] + krow[d]) * Ww[d];
  s += Wb[0];
  float m = mask[((size_t)q * TKn + k) * BBn + b];
  e[k] = (m == 1.0f) ? s : -1.0e6f;
  __syncthreads();
  red[tid] = e[tid];
  __syncthreads();
  for (int st = TKn / 2; st > 0; st >>= 1) {
    if (tid < st) red[tid] = fmaxf(red[tid], red[tid + st]);
    __syncthreads();
  }
  float mx = red[0];
  __syncthreads();
  float av = expf(e[tid] - mx);
  e[tid] = av;
  red[tid] = av;
  __syncthreads();
  for (int st = TKn / 2; st > 0; st >>= 1) {
    if (tid < st) red[tid] += red[tid + st];
    __syncthreads();
  }
  float inv = 1.0f / red[0];
  __syncthreads();
  for (int d = tid; d < DD; d += TKn) {
    float acc = 0.0f;
    for (int kk = 0; kk < TKn; ++kk)
      acc += e[kk] * vp[((size_t)kk * BBn + b) * DD + d];
    ctx[((size_t)q * BBn + b) * DD + d] = acc * inv;
  }
}

__global__ void concat_kernel(const float* __restrict__ x,
                              const float* __restrict__ ctx,
                              float* __restrict__ xcat, int n) {
  int i = blockIdx.x * blockDim.x + threadIdx.x;
  if (i >= n) return;
  int r = i / (2 * DD), c = i - r * (2 * DD);
  xcat[i] = (c < DD) ? x[(size_t)r * DD + c] : ctx[(size_t)r * DD + (c - DD)];
}

extern "C" void kernel_launch(void* const* d_in, const int* in_sizes, int n_in,
                              void* d_out, int out_size, void* d_ws,
                              size_t ws_size, hipStream_t stream) {
  (void)in_sizes; (void)n_in; (void)out_size; (void)ws_size;
  const int*   inputs = (const int*)d_in[0];
  const float* h0     = (const float*)d_in[1];
  const float* c0     = (const float*)d_in[2];
  const float* enc    = (const float*)d_in[3];
  const float* mask   = (const float*)d_in[4];
  const float* emb    = (const float*)d_in[5];
  const float* WihS   = (const float*)d_in[6];
  const float* WhhS   = (const float*)d_in[7];
  const float* bihS   = (const float*)d_in[8];
  const float* bhhS   = (const float*)d_in[9];
  const float* Wih_f  = (const float*)d_in[10];
  const float* Whh_f  = (const float*)d_in[11];
  const float* bih_f  = (const float*)d_in[12];
  const float* bhh_f  = (const float*)d_in[13];
  const float* Qw = (const float*)d_in[14]; const float* Qb = (const float*)d_in[15];
  const float* Kw = (const float*)d_in[16]; const float* Kb = (const float*)d_in[17];
  const float* Vw = (const float*)d_in[18]; const float* Vb = (const float*)d_in[19];
  const float* Ww = (const float*)d_in[20]; const float* Wb = (const float*)d_in[21];
  const float* fcw = (const float*)d_in[22]; const float* fcb = (const float*)d_in[23];

  float* out_logits = (float*)d_out;                       // [TQ,B,V]
  float* out_h = out_logits + (size_t)TQn * BBn * VV;      // [L,B,D]
  float* out_c = out_h + (size_t)3 * BBn * DD;             // [L,B,D]

  // ---- workspace bump allocator ----
  char* base = (char*)d_ws;
  size_t off = 0;
  auto alloc = [&](size_t bytes) -> void* {
    void* p = base + off;
    off += (bytes + 255) & ~(size_t)255;
    return p;
  };
  float* x    = (float*)alloc(sizeof(float) * MR * DD);
  float* xg   = (float*)alloc(sizeof(float) * MR * G4);
  float* qp   = (float*)alloc(sizeof(float) * MR * DD);
  float* kp   = (float*)alloc(sizeof(float) * MR * DD);
  float* vp   = (float*)alloc(sizeof(float) * MR * DD);
  float* ctx  = (float*)alloc(sizeof(float) * MR * DD);
  float* xcat = (float*)alloc(sizeof(float) * MR * 2 * DD);
  float* yv   = (float*)alloc(sizeof(float) * MR * DD);
  float* Gbuf = (float*)alloc(sizeof(float) * BBn * G4);
  float* hbuf = (float*)alloc(sizeof(float) * BBn * DD);
  float* cbuf = (float*)alloc(sizeof(float) * BBn * DD);
  float* biasbuf = (float*)alloc(sizeof(float) * G4);
  unsigned short* x_bf    = (unsigned short*)alloc(2ull * MR * DD);
  unsigned short* enc_bf  = (unsigned short*)alloc(2ull * MR * DD);
  unsigned short* xcat_bf = (unsigned short*)alloc(2ull * MR * 2 * DD);
  unsigned short* y_bf    = (unsigned short*)alloc(2ull * MR * DD);
  unsigned short* h_bf    = (unsigned short*)alloc(2ull * BBn * DD);
  unsigned short* wihS_bf = (unsigned short*)alloc(2ull * 2 * G4 * DD);
  unsigned short* whhS_bf = (unsigned short*)alloc(2ull * 2 * G4 * DD);
  unsigned short* wihf_bf = (unsigned short*)alloc(2ull * G4 * 2 * DD);
  unsigned short* whhf_bf = (unsigned short*)alloc(2ull * G4 * DD);
  unsigned short* qw_bf   = (unsigned short*)alloc(2ull * DD * DD);
  unsigned short* kw_bf   = (unsigned short*)alloc(2ull * DD * DD);
  unsigned short* vw_bf   = (unsigned short*)alloc(2ull * DD * DD);
  unsigned short* fcw_bf  = (unsigned short*)alloc(2ull * VV * DD);

  auto cvt = [&](const float* s, unsigned short* d, int n) {
    cvt_bf16_kernel<<<dim3((n + 255) / 256), dim3(256), 0, stream>>>(s, d, n);
  };
  // small/recurrent GEMM (one tile per wave, fused Cin)
  auto gemm = [&](const unsigned short* A, const unsigned short* W,
                  const float* Cin, const float* bias, float* C,
                  int M, int N, int K) {
    int tiles = (M / 16) * (N / 16);
    wmma_gemm<<<dim3((tiles + 3) / 4), dim3(128), 0, stream>>>(
        A, W, Cin, bias, C, M, N, K, tiles);
  };
  // large GEMM (32x64 per wave register blocking)
  auto gemm_blk = [&](const unsigned short* A, const unsigned short* W,
                      const float* bias, float* C, int M, int N, int K) {
    int blks = (M / 32) * (N / 64);
    wmma_gemm_blk<<<dim3((blks + 3) / 4), dim3(128), 0, stream>>>(
        A, W, bias, C, M, N, K, blks);
  };

  // ---- bf16 weight / encoder conversions (per call; deterministic) ----
  cvt(WihS, wihS_bf, 2 * G4 * DD);
  cvt(WhhS, whhS_bf, 2 * G4 * DD);
  cvt(Wih_f, wihf_bf, G4 * 2 * DD);
  cvt(Whh_f, whhf_bf, G4 * DD);
  cvt(Qw, qw_bf, DD * DD);
  cvt(Kw, kw_bf, DD * DD);
  cvt(Vw, vw_bf, DD * DD);
  cvt(fcw, fcw_bf, VV * DD);
  cvt(enc, enc_bf, MR * DD);

  // ---- embedding ----
  embed_kernel<<<dim3((MR * DD + 255) / 256), dim3(256), 0, stream>>>(
      inputs, emb, x, MR * DD);

  dim3 bd256(256), g8k((BBn * DD + 255) / 256);

  // ---- stacked LSTM layers 0,1 with residual (in-place on x) ----
  for (int l = 0; l < 2; ++l) {
    add2_kernel<<<dim3((G4 + 255) / 256), bd256, 0, stream>>>(
        bihS + (size_t)l * G4, bhhS + (size_t)l * G4, biasbuf, G4);
    cvt(x, x_bf, MR * DD);
    gemm_blk(x_bf, wihS_bf + (size_t)l * G4 * DD, biasbuf, xg, MR, G4, DD);
    init_hc_kernel<<<g8k, bd256, 0, stream>>>(h0, c0, l, hbuf, cbuf, h_bf);
    for (int t = 0; t < TQn; ++t) {
      // G = xg[t] + h @ Whh^T  (M=16 batch rows -> exactly one WMMA M tile)
      gemm(h_bf, whhS_bf + (size_t)l * G4 * DD, xg + (size_t)t * BBn * G4,
           nullptr, Gbuf, BBn, G4, DD);
      lstm_point_kernel<<<g8k, bd256, 0, stream>>>(Gbuf, hbuf, cbuf, h_bf,
                                                   x /*yout*/, x /*skip*/, t);
    }
    copy_hc_kernel<<<g8k, bd256, 0, stream>>>(hbuf, cbuf,
                                              out_h + (size_t)l * BBn * DD,
                                              out_c + (size_t)l * BBn * DD);
  }

  // ---- attention ----
  cvt(x, x_bf, MR * DD);
  gemm_blk(x_bf, qw_bf, Qb, qp, MR, DD, DD);
  gemm_blk(enc_bf, kw_bf, Kb, kp, MR, DD, DD);   // TK*B == MR
  gemm_blk(enc_bf, vw_bf, Vb, vp, MR, DD, DD);
  attn_kernel<<<dim3(MR), dim3(TKn), 0, stream>>>(qp, kp, vp, mask, Ww, Wb, ctx);

  // ---- final LSTM on concat(x, ctx) ----
  add2_kernel<<<dim3((G4 + 255) / 256), bd256, 0, stream>>>(bih_f, bhh_f,
                                                            biasbuf, G4);
  concat_kernel<<<dim3((MR * 2 * DD + 255) / 256), bd256, 0, stream>>>(
      x, ctx, xcat, MR * 2 * DD);
  cvt(xcat, xcat_bf, MR * 2 * DD);
  gemm_blk(xcat_bf, wihf_bf, biasbuf, xg, MR, G4, 2 * DD);
  init_hc_kernel<<<g8k, bd256, 0, stream>>>(h0, c0, 2, hbuf, cbuf, h_bf);
  for (int t = 0; t < TQn; ++t) {
    gemm(h_bf, whhf_bf, xg + (size_t)t * BBn * G4, nullptr, Gbuf, BBn, G4, DD);
    lstm_point_kernel<<<g8k, bd256, 0, stream>>>(Gbuf, hbuf, cbuf, h_bf,
                                                 yv, nullptr, t);
  }
  copy_hc_kernel<<<g8k, bd256, 0, stream>>>(hbuf, cbuf,
                                            out_h + (size_t)2 * BBn * DD,
                                            out_c + (size_t)2 * BBn * DD);

  // ---- logits = y @ fcw^T + fcb  (2048 x 32000 x 512, writes d_out) ----
  cvt(yv, y_bf, MR * DD);
  gemm_blk(y_bf, fcw_bf, fcb, out_logits, MR, VV, DD);
}
